// SGMTSEncoder_43052752175731
// MI455X (gfx1250) — matrix operand order repaired and checked by reading, not compiled, not run on hardware
//
#include <hip/hip_runtime.h>

typedef __bf16 bf16_t;
typedef bf16_t v16bf __attribute__((ext_vector_type(16)));
typedef float  v8f   __attribute__((ext_vector_type(8)));
typedef unsigned int u32x4 __attribute__((ext_vector_type(4)));
typedef int i32x4 __attribute__((ext_vector_type(4)));
typedef int i32x8 __attribute__((ext_vector_type(8)));

#define B_  8
#define C_  3
#define HI_ 448
#define WI_ 448
#define NH_ 28
#define P_  784
#define BP_ 6272
#define DP_ 768
#define DH_ 384
#define DF_ 256
#define DS_ 16
#define DL_ 896
#define E_  1512

__device__ __forceinline__ bf16_t f2bf(float x) {
  union { float f; unsigned u; } v; v.f = x;
  unsigned r = v.u + 0x7FFFu + ((v.u >> 16) & 1u);
  unsigned short h = (unsigned short)(r >> 16);
  return __builtin_bit_cast(bf16_t, h);
}

__device__ __forceinline__ float apply_act(float x, int act) {
  if (act == 1) return 0.5f * x * (1.0f + erff(x * 0.70710678118654752f)); // exact GELU
  if (act == 2) return (x > 20.0f) ? x : log1pf(__expf(x));               // softplus
  return x;
}

// ---------------------------------------------------------------------------
// TDM: DMA a 2-D bf16 tile (tile_w x tile_h, row stride = row_stride elements)
// from global memory into LDS at byte offset lds_off. D# per CDNA5 ISA §8.3/8.4.
// ---------------------------------------------------------------------------
__device__ __forceinline__ void tdm_load_tile_bf16(
    unsigned lds_off, const bf16_t* gptr,
    unsigned tile_w, unsigned tile_h,
    unsigned tensor_w, unsigned tensor_h, unsigned row_stride)
{
  unsigned long long ga = (unsigned long long)(const void*)gptr;
  u32x4 g0;
  g0[0] = 1u;                                                // count=1, user D#, no gather
  g0[1] = lds_off;                                           // lds_addr (bytes)
  g0[2] = (unsigned)(ga & 0xFFFFFFFFu);                      // global_addr[31:0]
  g0[3] = (unsigned)((ga >> 32) & 0x01FFFFFFu) | (2u << 30); // global_addr[56:32], type=2
  i32x8 g1;
  g1[0] = (int)(1u << 16);                                   // data_size=1 (2B); wg_mask=0
  g1[1] = (int)((tensor_w & 0xFFFFu) << 16);                 // tensor_dim0[15:0]
  g1[2] = (int)(((tensor_w >> 16) & 0xFFFFu) | ((tensor_h & 0xFFFFu) << 16));
  g1[3] = (int)(((tensor_h >> 16) & 0xFFFFu) | ((tile_w & 0xFFFFu) << 16));
  g1[4] = (int)(tile_h & 0xFFFFu);                           // tile_dim1 (tile_dim2=0)
  g1[5] = (int)row_stride;                                   // tensor_dim0_stride[31:0]
  g1[6] = 0;                                                 // stride0[47:32], stride1 lo
  g1[7] = 0;
  i32x4 z4 = {0, 0, 0, 0};
#if defined(__clang_major__) && (__clang_major__ >= 23)
  i32x8 z8 = {0, 0, 0, 0, 0, 0, 0, 0};
  __builtin_amdgcn_tensor_load_to_lds(g0, g1, z4, z4, z8, 0);
#else
  __builtin_amdgcn_tensor_load_to_lds(g0, g1, z4, z4, 0);
#endif
}

__device__ __forceinline__ unsigned lds_off_of(const void* p) {
  return (unsigned)(size_t)p;
}

// ---------------------------------------------------------------------------
// bf16 WMMA GEMM:  C[M,N] = act(A[M,K] @ Bw[N,K]^T + bias)
// block tile 64x64, 256 threads = 8 waves, each wave: 16(M) x 32(N) via 2 WMMAs
// Tiles staged into LDS by the Tensor Data Mover, double-buffered.
// ---------------------------------------------------------------------------
#define TM 64
#define TN 64
#define TK 32

__global__ __launch_bounds__(256) void gemm_bf16_wmma(
    const bf16_t* __restrict__ A, const bf16_t* __restrict__ Bw,
    const float* __restrict__ bias,
    float* __restrict__ Cf, bf16_t* __restrict__ Cb,
    int M, int N, int K, int act)
{
  __shared__ __align__(16) bf16_t As[2][TM * TK];
  __shared__ __align__(16) bf16_t Bs[2][TN * TK];

  const int tid  = threadIdx.x;
  const int lane = tid & 31;
  const int m0 = blockIdx.x * TM;
  const int n0 = blockIdx.y * TN;
  const int wm = ((tid >> 5) >> 1) * 16;   // 0,16,32,48
  const int wn = ((tid >> 5) & 1) * 32;    // 0,32

  v8f acc0 = {}; v8f acc1 = {};

  const int ntiles = K / TK;
  const bool issuer = (tid < 32);          // wave 0 drives the TDM

  if (issuer) {
    tdm_load_tile_bf16(lds_off_of(&As[0][0]), A  + (size_t)m0 * K,      TK, TM, (unsigned)K, (unsigned)M, (unsigned)K);
    tdm_load_tile_bf16(lds_off_of(&Bs[0][0]), Bw + (size_t)n0 * K,      TK, TN, (unsigned)K, (unsigned)N, (unsigned)K);
    if (ntiles > 1) {
      tdm_load_tile_bf16(lds_off_of(&As[1][0]), A  + (size_t)m0 * K + TK, TK, TM, (unsigned)K, (unsigned)M, (unsigned)K);
      tdm_load_tile_bf16(lds_off_of(&Bs[1][0]), Bw + (size_t)n0 * K + TK, TK, TN, (unsigned)K, (unsigned)N, (unsigned)K);
    }
  }

  for (int it = 0; it < ntiles; ++it) {
    if (issuer) {
      if (it + 1 < ntiles) __builtin_amdgcn_s_wait_tensorcnt(2);  // tile 'it' complete
      else                 __builtin_amdgcn_s_wait_tensorcnt(0);  // drain on last tile
    }
    __syncthreads();   // publish DMA'd buffer to all waves

    const int buf = it & 1;
    // A fragment (16x32): lane<16 rows 0..15 K=0..15; lane>=16 K=16..31
    v16bf a{};
    {
      const int row   = lane & 15;
      const int hbase = (lane >> 4) << 4;
      const bf16_t* ap = &As[buf][(wm + row) * TK];
#pragma unroll
      for (int e = 0; e < 16; ++e) {
        int kk = e + hbase;
        a[e] = ap[kk];
      }
    }
    // B fragments (32x16): lane<16 K=0..15, lane>=16 K=16..31; col = lane&15
    v16bf b0{}, b1{};
    {
      const int col  = lane & 15;
      const int kadd = (lane >> 4) << 4;
      const bf16_t* bp0 = &Bs[buf][(wn + col) * TK];
      const bf16_t* bp1 = &Bs[buf][(wn + 16 + col) * TK];
#pragma unroll
      for (int e = 0; e < 16; ++e) {
        b0[e] = bp0[e + kadd];
        b1[e] = bp1[e + kadd];
      }
    }
    acc0 = __builtin_amdgcn_wmma_f32_16x16x32_bf16(false, a, false, b0, (short)0, acc0, false, false);
    acc1 = __builtin_amdgcn_wmma_f32_16x16x32_bf16(false, a, false, b1, (short)0, acc1, false, false);

    __syncthreads();   // all waves done reading buf before DMA overwrites it
    if (issuer && (it + 2) < ntiles) {
      const int kn = (it + 2) * TK;
      tdm_load_tile_bf16(lds_off_of(&As[buf][0]), A  + (size_t)m0 * K + kn, TK, TM, (unsigned)K, (unsigned)M, (unsigned)K);
      tdm_load_tile_bf16(lds_off_of(&Bs[buf][0]), Bw + (size_t)n0 * K + kn, TK, TN, (unsigned)K, (unsigned)N, (unsigned)K);
    }
  }

  // C/D layout: VGPR r -> row r (lanes 0-15) / row r+8 (lanes 16-31); col = lane&15
  const int lane16 = lane & 15;
  const int rsel   = (lane >> 4) << 3;
#pragma unroll
  for (int r = 0; r < 8; ++r) {
    int row = m0 + wm + rsel + r;
    int c0  = n0 + wn + lane16;
    int c1  = c0 + 16;
    float x0 = apply_act(acc0[r] + bias[c0], act);
    float x1 = apply_act(acc1[r] + bias[c1], act);
    if (Cf) { Cf[(size_t)row * N + c0] = x0;        Cf[(size_t)row * N + c1] = x1; }
    if (Cb) { Cb[(size_t)row * N + c0] = f2bf(x0);  Cb[(size_t)row * N + c1] = f2bf(x1); }
  }
}

// ---------------------------------------------------------------------------
__global__ void cvt_bf16_kernel(const float* __restrict__ in, bf16_t* __restrict__ out, int n) {
  int i = blockIdx.x * 256 + threadIdx.x;
  if (i < n) out[i] = f2bf(in[i]);
}

// images (B,C,H,W) -> patches (B,P, c*256+py*16+px) in bf16
__global__ void patchify_kernel(const float* __restrict__ img, bf16_t* __restrict__ out) {
  long idx = (long)blockIdx.x * 256 + threadIdx.x;
  if (idx >= (long)B_ * P_ * DP_) return;
  int f = (int)(idx % DP_);
  long r = idx / DP_;
  int p = (int)(r % P_);
  int b = (int)(r / P_);
  int c  = f >> 8;
  int py = (f >> 4) & 15;
  int px = f & 15;
  int i = p / NH_, j = p % NH_;
  int hh = i * 16 + py, ww = j * 16 + px;
  out[idx] = f2bf(img[(((size_t)b * C_ + c) * HI_ + hh) * WI_ + ww]);
}

// LayerNorm over width 256, one row per block
__global__ __launch_bounds__(256) void layernorm_kernel(
    const float* __restrict__ X, const float* __restrict__ g, const float* __restrict__ bb,
    float* __restrict__ Y)
{
  __shared__ float red[256];
  const int row = blockIdx.x, t = threadIdx.x;
  float x = X[(size_t)row * DF_ + t];
  red[t] = x; __syncthreads();
  for (int o = 128; o; o >>= 1) { if (t < o) red[t] += red[t + o]; __syncthreads(); }
  float m = red[0] * (1.0f / DF_);
  __syncthreads();
  float d = x - m;
  red[t] = d * d; __syncthreads();
  for (int o = 128; o; o >>= 1) { if (t < o) red[t] += red[t + o]; __syncthreads(); }
  float v = red[0] * (1.0f / DF_);
  Y[(size_t)row * DF_ + t] = d * rsqrtf(v + 1e-5f) * g[t] + bb[t];
}

// out[b][n] = sum_k x[b][k] * W[n][k]   (N=256 threads, grid=B)
__global__ __launch_bounds__(256) void matvec_kernel(
    const float* __restrict__ x, const float* __restrict__ W, float* __restrict__ out, int K)
{
  int b = blockIdx.x, n = threadIdx.x;
  const float* xr = x + (size_t)b * K;
  const float* wr = W + (size_t)n * K;
  float s = 0.f;
  for (int k = 0; k < K; ++k) s += xr[k] * wr[k];
  out[b * DF_ + n] = s;
}

// row-wise L2 normalize (8 rows of 256)
__global__ __launch_bounds__(256) void norm_rows_kernel(const float* __restrict__ in, float* __restrict__ out) {
  __shared__ float red[256];
  int b = blockIdx.x, t = threadIdx.x;
  float x = in[b * DF_ + t];
  red[t] = x * x; __syncthreads();
  for (int o = 128; o; o >>= 1) { if (t < o) red[t] += red[t + o]; __syncthreads(); }
  float n = fmaxf(sqrtf(red[0]), 1e-12f);
  out[b * DF_ + t] = x / n;
}

// fnorm = feats/||feats||_row ; sem[b,p] = fnorm . gn[b]
__global__ __launch_bounds__(256) void rownorm_sem_kernel(
    const float* __restrict__ feats, const float* __restrict__ gn,
    float* __restrict__ fnorm, float* __restrict__ sem)
{
  __shared__ float red[256];
  int row = blockIdx.x;       // b*P + p
  int b = row / P_;
  int t = threadIdx.x;
  float x = feats[(size_t)row * DF_ + t];
  red[t] = x * x; __syncthreads();
  for (int o = 128; o; o >>= 1) { if (t < o) red[t] += red[t + o]; __syncthreads(); }
  float n = fmaxf(sqrtf(red[0]), 1e-12f);
  __syncthreads();
  float f = x / n;
  fnorm[(size_t)row * DF_ + t] = f;
  red[t] = f * gn[b * DF_ + t]; __syncthreads();
  for (int o = 128; o; o >>= 1) { if (t < o) red[t] += red[t + o]; __syncthreads(); }
  if (t == 0) sem[row] = red[0];
}

__device__ __forceinline__ void edge_uv(int e, int& u, int& v) {
  if (e < (NH_ * (NH_ - 1))) { int i = e / (NH_ - 1), j = e % (NH_ - 1); u = i * NH_ + j; v = u + 1; }
  else { int r = e - NH_ * (NH_ - 1); int i = r / NH_, j = r % NH_; u = i * NH_ + j; v = u + NH_; }
}

// one wave per edge: w = fn[u].fn[v] + 0.5*sem[u]*sem[v]
__global__ __launch_bounds__(256) void edge_weight_kernel(
    const float* __restrict__ fnorm, const float* __restrict__ sem, float* __restrict__ wts)
{
  int gid = blockIdx.x * 8 + (threadIdx.x >> 5);
  if (gid >= B_ * E_) return;
  int b = gid / E_, e = gid % E_;
  int u, v; edge_uv(e, u, v);
  int lane = threadIdx.x & 31;
  const float* fu = fnorm + ((size_t)b * P_ + u) * DF_;
  const float* fv = fnorm + ((size_t)b * P_ + v) * DF_;
  float s = 0.f;
  for (int k = lane; k < DF_; k += 32) s += fu[k] * fv[k];
  for (int o = 16; o; o >>= 1) s += __shfl_xor(s, o, 32);
  if (lane == 0) wts[gid] = s + 0.5f * sem[b * P_ + u] * sem[b * P_ + v];
}

// one block per sample: bitonic sort (desc w) -> Kruskal union-find -> BFS
__global__ __launch_bounds__(1024) void build_structure_kernel(
    const float* __restrict__ wts, const float* __restrict__ sem,
    int* __restrict__ bfs, int* __restrict__ par, int* __restrict__ inv)
{
  __shared__ float key[2048];
  __shared__ int   val[2048];
  __shared__ int   uf[P_];
  __shared__ int   deg[P_];
  __shared__ int   nbr[P_ * 4];
  __shared__ int   parent_of[P_];
  __shared__ int   order[P_];

  const int b = blockIdx.x, t = threadIdx.x;
  for (int i = t; i < 2048; i += 1024) {
    if (i < E_) { key[i] = -wts[b * E_ + i]; val[i] = i; }   // ascending -w == descending w
    else        { key[i] = 3.4e38f;          val[i] = -1; }
  }
  for (int i = t; i < P_; i += 1024) { uf[i] = i; deg[i] = 0; parent_of[i] = -1; }
  __syncthreads();

  for (int k = 2; k <= 2048; k <<= 1) {
    for (int j = k >> 1; j > 0; j >>= 1) {
      int i = t;
      for (int rep = 0; rep < 2; ++rep, i += 1024) {
        int ixj = i ^ j;
        if (ixj > i) {
          bool up = ((i & k) == 0);
          bool sw = up ? (key[i] > key[ixj]) : (key[i] < key[ixj]);
          if (sw) {
            float fk = key[i]; key[i] = key[ixj]; key[ixj] = fk;
            int vv = val[i]; val[i] = val[ixj]; val[ixj] = vv;
          }
        }
      }
      __syncthreads();
    }
  }

  if (t == 0) {
    int cnt = 0;
    for (int e = 0; e < E_ && cnt < P_ - 1; ++e) {
      int id = val[e];
      int u, v; edge_uv(id, u, v);
      int ru = u; while (uf[ru] != ru) { uf[ru] = uf[uf[ru]]; ru = uf[ru]; }
      int rv = v; while (uf[rv] != rv) { uf[rv] = uf[uf[rv]]; rv = uf[rv]; }
      if (ru != rv) {
        uf[ru] = rv;
        nbr[u * 4 + deg[u]] = v; deg[u]++;
        nbr[v * 4 + deg[v]] = u; deg[v]++;
        ++cnt;
      }
    }
    int root = 0; float best = sem[b * P_];
    for (int i = 1; i < P_; ++i) { float sv = sem[b * P_ + i]; if (sv > best) { best = sv; root = i; } }
    for (int i = 0; i < P_; ++i) uf[i] = 0;       // reuse as visited
    int head = 0, tail = 0;
    order[tail++] = root; uf[root] = 1;
    while (head < tail) {
      int u = order[head++];
      for (int q = 0; q < deg[u]; ++q) {
        int v = nbr[u * 4 + q];
        if (!uf[v]) { uf[v] = 1; parent_of[v] = u; order[tail++] = v; }
      }
    }
    for (int pos = 0; pos < P_; ++pos) { int o = order[pos]; bfs[b * P_ + pos] = o; inv[b * P_ + o] = pos; }
    for (int pos = 0; pos < P_; ++pos) {
      int o = order[pos]; int po = parent_of[o];
      par[b * P_ + pos] = (po < 0) ? -1 : inv[b * P_ + po];
    }
  }
}

// X[b,i,:] = feats[b,bfs[i],:] + sem[b,bfs[i]]*g_prime[b,:]  (f32 + bf16)
__global__ void build_X_kernel(
    const float* __restrict__ feats, const float* __restrict__ sem,
    const float* __restrict__ gprime, const int* __restrict__ bfs,
    float* __restrict__ X, bf16_t* __restrict__ Xb)
{
  long idx = (long)blockIdx.x * 256 + threadIdx.x;   // BP_*DF_ total
  int d = (int)(idx & (DF_ - 1));
  long r = idx >> 8;
  int i = (int)(r % P_), b = (int)(r / P_);
  int o = bfs[b * P_ + i];
  float x = feats[((size_t)b * P_ + o) * DF_ + d] + sem[b * P_ + o] * gprime[b * DF_ + d];
  X[idx] = x;
  Xb[idx] = f2bf(x);
}

// Bp/Cp: one row per 32-thread block (t<16 -> Bp, else Cp)
__global__ __launch_bounds__(32) void bpcp_kernel(
    const float* __restrict__ X,
    const float* __restrict__ BpW, const float* __restrict__ Bpb,
    const float* __restrict__ CpW, const float* __restrict__ Cpb,
    float* __restrict__ Bp, float* __restrict__ Cp)
{
  int row = blockIdx.x, t = threadIdx.x;
  const float* x = X + (size_t)row * DF_;
  const float* W = (t < 16) ? (BpW + (size_t)t * DF_) : (CpW + (size_t)(t - 16) * DF_);
  float s = (t < 16) ? Bpb[t] : Cpb[t - 16];
  for (int k = 0; k < DF_; ++k) s += x[k] * W[k];
  if (t < 16) Bp[(size_t)row * DS_ + t] = s;
  else        Cp[(size_t)row * DS_ + (t - 16)] = s;
}

// tree-structured selective scan: block = one (b,d), history private per lane in LDS
__global__ __launch_bounds__(16) void tree_scan_kernel(
    const float* __restrict__ X, const float* __restrict__ delta,
    const float* __restrict__ Bp, const float* __restrict__ Cp,
    const int* __restrict__ par, const float* __restrict__ A_log,
    const float* __restrict__ D, float* __restrict__ Y)
{
  __shared__ float H[P_ * DS_];   // 49 KB, per-lane-private history
  const int b = blockIdx.x, d = blockIdx.y, s = threadIdx.x;
  const float A  = -__expf(A_log[d * DS_ + s]);
  const float Dv = D[d];
  const size_t base = (size_t)b * P_;
  for (int i = 0; i < P_; ++i) {
    if (i + 1 < P_) {   // hide next step's global latency
      __builtin_prefetch(&delta[(base + i + 1) * DF_ + d], 0, 0);
      __builtin_prefetch(&X[(base + i + 1) * DF_ + d], 0, 0);
      __builtin_prefetch(&Bp[(base + i + 1) * DS_ + s], 0, 0);
    }
    float dlt = delta[(base + i) * DF_ + d];
    float xv  = X[(base + i) * DF_ + d];
    float u   = dlt * xv * Bp[(base + i) * DS_ + s];
    int p = par[base + i];
    float hp = (p >= 0) ? H[p * DS_ + s] : 0.0f;
    float h = __expf(dlt * A) * hp + u;
    H[i * DS_ + s] = h;
    float y = h * Cp[(base + i) * DS_ + s];
    for (int o = 8; o; o >>= 1) y += __shfl_xor(y, o, 16);
    if (s == 0) Y[(base + i) * DF_ + d] = y + Dv * xv;
  }
}

// raster reorder + bf16 for final GEMM
__global__ void reorder_kernel(const float* __restrict__ Y, const int* __restrict__ inv,
                               bf16_t* __restrict__ Yrb)
{
  long idx = (long)blockIdx.x * 256 + threadIdx.x;   // BP_*DF_
  int d = (int)(idx & (DF_ - 1));
  long r = idx >> 8;
  int p = (int)(r % P_), b = (int)(r / P_);
  int i = inv[b * P_ + p];
  Yrb[idx] = f2bf(Y[((size_t)b * P_ + i) * DF_ + d]);
}

// ---------------------------------------------------------------------------
extern "C" void kernel_launch(void* const* d_in, const int* in_sizes, int n_in,
                              void* d_out, int out_size, void* d_ws, size_t ws_size,
                              hipStream_t stream)
{
  (void)in_sizes; (void)n_in; (void)out_size; (void)ws_size;
  const float* images  = (const float*)d_in[0];
  const float* g_task  = (const float*)d_in[1];
  const float* pc_W1   = (const float*)d_in[2];
  const float* pc_b1   = (const float*)d_in[3];
  const float* pc_W2   = (const float*)d_in[4];
  const float* pc_b2   = (const float*)d_in[5];
  const float* pc_ln_g = (const float*)d_in[6];
  const float* pc_ln_b = (const float*)d_in[7];
  const float* lang_W  = (const float*)d_in[8];
  const float* gprW    = (const float*)d_in[9];
  const float* A_log   = (const float*)d_in[10];
  const float* D_param = (const float*)d_in[11];
  const float* Bp_W    = (const float*)d_in[12];
  const float* Bp_b    = (const float*)d_in[13];
  const float* Cp_W    = (const float*)d_in[14];
  const float* Cp_b    = (const float*)d_in[15];
  const float* Wd_W    = (const float*)d_in[16];
  const float* Wd_b    = (const float*)d_in[17];
  const float* out_W   = (const float*)d_in[18];
  const float* out_b   = (const float*)d_in[19];
  const float* on_g    = (const float*)d_in[20];
  const float* on_b    = (const float*)d_in[21];
  float* out = (float*)d_out;

  uint8_t* ws = (uint8_t*)d_ws;
  size_t off = 0;
  auto alloc = [&](size_t bytes) { size_t o = off; off += (bytes + 255) & ~(size_t)255; return o; };

  size_t o_patch = alloc((size_t)BP_ * DP_ * 2);   // bf16 patches; later reused as f32 delta
  size_t o_W1b   = alloc((size_t)DH_ * DP_ * 2);
  size_t o_W2b   = alloc((size_t)DF_ * DH_ * 2);
  size_t o_Wdb   = alloc((size_t)DF_ * DF_ * 2);
  size_t o_oWb   = alloc((size_t)DF_ * DF_ * 2);
  size_t o_h1b   = alloc((size_t)BP_ * DH_ * 2);   // bf16 h1; later reused as Xb
  size_t o_fpre  = alloc((size_t)BP_ * DF_ * 4);   // pre-LN feats; later reused as Yrb
  size_t o_feats = alloc((size_t)BP_ * DF_ * 4);
  size_t o_fnorm = alloc((size_t)BP_ * DF_ * 4);   // later reused as Y
  size_t o_X     = alloc((size_t)BP_ * DF_ * 4);   // later reused as out-pre-LN
  size_t o_ggate = alloc((size_t)B_ * DF_ * 4);
  size_t o_gn    = alloc((size_t)B_ * DF_ * 4);
  size_t o_gpr   = alloc((size_t)B_ * DF_ * 4);
  size_t o_sem   = alloc((size_t)B_ * P_ * 4);
  size_t o_wts   = alloc((size_t)B_ * E_ * 4);
  size_t o_bfs   = alloc((size_t)B_ * P_ * 4);
  size_t o_par   = alloc((size_t)B_ * P_ * 4);
  size_t o_inv   = alloc((size_t)B_ * P_ * 4);
  size_t o_Bp    = alloc((size_t)BP_ * DS_ * 4);
  size_t o_Cp    = alloc((size_t)BP_ * DS_ * 4);

  bf16_t* patch = (bf16_t*)(ws + o_patch);
  bf16_t* W1b   = (bf16_t*)(ws + o_W1b);
  bf16_t* W2b   = (bf16_t*)(ws + o_W2b);
  bf16_t* Wdb   = (bf16_t*)(ws + o_Wdb);
  bf16_t* oWb   = (bf16_t*)(ws + o_oWb);
  bf16_t* h1b   = (bf16_t*)(ws + o_h1b);
  float*  fpre  = (float*)(ws + o_fpre);
  float*  feats = (float*)(ws + o_feats);
  float*  fnorm = (float*)(ws + o_fnorm);
  float*  X     = (float*)(ws + o_X);
  float*  ggate = (float*)(ws + o_ggate);
  float*  gn    = (float*)(ws + o_gn);
  float*  gpr   = (float*)(ws + o_gpr);
  float*  sem   = (float*)(ws + o_sem);
  float*  wts   = (float*)(ws + o_wts);
  int*    bfs   = (int*)(ws + o_bfs);
  int*    par   = (int*)(ws + o_par);
  int*    inv   = (int*)(ws + o_inv);
  float*  Bp    = (float*)(ws + o_Bp);
  float*  Cp    = (float*)(ws + o_Cp);
  // aliases (safe by pipeline ordering)
  float*  delta = (float*)(ws + o_patch);   // after GEMM1 consumed patches
  bf16_t* Xb    = (bf16_t*)(ws + o_h1b);    // after GEMM2 consumed h1
  float*  Y     = (float*)(ws + o_fnorm);   // after edge weights consumed fnorm
  bf16_t* Yrb   = (bf16_t*)(ws + o_fpre);   // after LN consumed fpre
  float*  opre  = (float*)(ws + o_X);       // after scan consumed X

  // 1) weight conversions
  cvt_bf16_kernel<<<(DH_ * DP_ + 255) / 256, 256, 0, stream>>>(pc_W1, W1b, DH_ * DP_);
  cvt_bf16_kernel<<<(DF_ * DH_ + 255) / 256, 256, 0, stream>>>(pc_W2, W2b, DF_ * DH_);
  cvt_bf16_kernel<<<(DF_ * DF_ + 255) / 256, 256, 0, stream>>>(Wd_W, Wdb, DF_ * DF_);
  cvt_bf16_kernel<<<(DF_ * DF_ + 255) / 256, 256, 0, stream>>>(out_W, oWb, DF_ * DF_);

  // 2) patchify
  patchify_kernel<<<(B_ * P_ * DP_ + 255) / 256, 256, 0, stream>>>(images, patch);

  // 3) h1 = gelu(patch @ W1^T + b1)   [6272 x 768] x [768 x 384]
  gemm_bf16_wmma<<<dim3(BP_ / TM, DH_ / TN), 256, 0, stream>>>(
      patch, W1b, pc_b1, nullptr, h1b, BP_, DH_, DP_, 1);

  // 4) feats_pre = h1 @ W2^T + b2     [6272 x 384] x [384 x 256]
  gemm_bf16_wmma<<<dim3(BP_ / TM, DF_ / TN), 256, 0, stream>>>(
      h1b, W2b, pc_b2, fpre, nullptr, BP_, DF_, DH_, 0);

  // 5) feats = LN(feats_pre)
  layernorm_kernel<<<BP_, 256, 0, stream>>>(fpre, pc_ln_g, pc_ln_b, feats);

  // 6) g_gate / g_prime ; gn
  matvec_kernel<<<B_, 256, 0, stream>>>(g_task, lang_W, ggate, DL_);
  matvec_kernel<<<B_, 256, 0, stream>>>(g_task, gprW, gpr, DL_);
  norm_rows_kernel<<<B_, 256, 0, stream>>>(ggate, gn);

  // 7) fnorm + sem
  rownorm_sem_kernel<<<BP_, 256, 0, stream>>>(feats, gn, fnorm, sem);

  // 8) edge weights
  edge_weight_kernel<<<(B_ * E_ + 7) / 8, 256, 0, stream>>>(fnorm, sem, wts);

  // 9) MST + BFS
  build_structure_kernel<<<B_, 1024, 0, stream>>>(wts, sem, bfs, par, inv);

  // 10) X (bfs order) + bf16 copy
  build_X_kernel<<<BP_, 256, 0, stream>>>(feats, sem, gpr, bfs, X, Xb);

  // 11) delta = softplus(X @ Wd^T + Wd_b)
  gemm_bf16_wmma<<<dim3(BP_ / TM, DF_ / TN), 256, 0, stream>>>(
      Xb, Wdb, Wd_b, delta, nullptr, BP_, DF_, DF_, 2);

  // 12) Bp / Cp
  bpcp_kernel<<<BP_, 32, 0, stream>>>(X, Bp_W, Bp_b, Cp_W, Cp_b, Bp, Cp);

  // 13) tree scan (Y in bfs order, includes + D*X)
  tree_scan_kernel<<<dim3(B_, DF_), 16, 0, stream>>>(X, delta, Bp, Cp, par, A_log, D_param, Y);

  // 14) raster reorder -> bf16
  reorder_kernel<<<BP_, 256, 0, stream>>>(Y, inv, Yrb);

  // 15) out_pre = Yr @ out_W^T + out_b
  gemm_bf16_wmma<<<dim3(BP_ / TM, DF_ / TN), 256, 0, stream>>>(
      Yrb, oWb, out_b, opre, nullptr, BP_, DF_, DF_, 0);

  // 16) final LN -> d_out
  layernorm_kernel<<<BP_, 256, 0, stream>>>(opre, on_g, on_b, out);
}
